// STSGCM_2370821947438
// MI455X (gfx1250) — matrix operand order repaired and checked by reading, not compiled, not run on hardware
//
#include <hip/hip_runtime.h>
#include <hip/hip_bf16.h>
#include <cstdint>

typedef __bf16 bf16_t;
typedef __attribute__((ext_vector_type(16))) __bf16 v16bf;
typedef __attribute__((ext_vector_type(8)))  float  v8f;

#define NV    3000          // real vertices (3 * N_VERT)
#define MPAD  3008          // padded vertex count (multiple of 64)
#define KPAD  3008          // padded contraction dim
#define NCOL  2048          // B * IN_DIM = 32 * 64
#define ODIM  64
#define NKSTEPS (KPAD / 32) // 94
#define WS_ALIGN(x) (((x) + 255) & ~(size_t)255)

// ---------------- conversion / init kernels ----------------

__global__ void k_cvt_adj(const float* __restrict__ adj, bf16_t* __restrict__ adjb) {
    long long i = (long long)blockIdx.x * blockDim.x + threadIdx.x;
    if (i >= (long long)MPAD * KPAD) return;
    int r = (int)(i / KPAD), c = (int)(i % KPAD);
    float v = (r < NV && c < NV) ? adj[(long long)r * NV + c] : 0.0f;
    adjb[i] = (bf16_t)v;
}

// x (3000,2048) f32 -> hT (2048, 3008) bf16, K-contiguous; LDS-tiled transpose
__global__ __launch_bounds__(256)
void k_cvt_xT(const float* __restrict__ x, bf16_t* __restrict__ hT) {
    __shared__ float tile[32][33];
    const int v0 = blockIdx.x * 32;   // vertex tile (0..93)
    const int c0 = blockIdx.y * 32;   // feature tile (0..63)
    for (int rr = threadIdx.y; rr < 32; rr += 8) {
        int v = v0 + rr;
        tile[rr][threadIdx.x] = (v < NV) ? x[(long long)v * NCOL + c0 + threadIdx.x] : 0.0f;
    }
    __syncthreads();
    for (int rr = threadIdx.y; rr < 32; rr += 8) {
        hT[(long long)(c0 + rr) * MPAD + v0 + threadIdx.x] = (bf16_t)tile[threadIdx.x][rr];
    }
}

// Wt layout per layer: [128 j][64 c] row-major (B-operand fragment friendly)
__global__ void k_cvt_w(const float* __restrict__ W0, const float* __restrict__ W1,
                        const float* __restrict__ W2, bf16_t* __restrict__ wt) {
    int i = blockIdx.x * blockDim.x + threadIdx.x;
    if (i >= 3 * 128 * ODIM) return;
    int l = i / (128 * ODIM);
    int rem = i % (128 * ODIM);
    int j = rem / ODIM, c = rem % ODIM;
    const float* W = (l == 0) ? W0 : (l == 1) ? W1 : W2;
    wt[i] = (bf16_t)W[c * 128 + j];
}

// zero the padded vertices 3000..3007 of a transposed activation buffer
__global__ void k_zero_padT(bf16_t* __restrict__ hT) {
    int i = blockIdx.x * blockDim.x + threadIdx.x;
    if (i < NCOL * (MPAD - NV)) {
        int c = i >> 3, p = i & 7;
        hT[(long long)c * MPAD + NV + p] = (bf16_t)0.0f;
    }
}

// ---------------- fused layer kernel ----------------
// 128 threads = 4 waves; 64(M)x64(N) tile of  Hagg = adj @ h  (bf16 WMMA, f32 acc),
// staged via double-buffered global_load_async_to_lds_b128, then fused
// lr = Htile @ W + bias ; GLU ; transposed bf16 store ; running-max output.

union SMem {
    struct {
        bf16_t A[2][64][32];   // K-contiguous rows, octets in natural order -> 1 read/frag
        bf16_t Bt[2][64][32];  // [col][K], K-contiguous
    } s;                       // 16 KB staging (double buffered)
    float H[64][64];           // 16 KB aggregate tile (reuses staging space)
};

__global__ __launch_bounds__(128)
void k_gcn_layer(const bf16_t* __restrict__ adjb,
                 const bf16_t* __restrict__ hT,     // [2048][3008] bf16
                 const bf16_t* __restrict__ wt,     // [128][64] bf16
                 const float*  __restrict__ bias,   // [128] f32
                 bf16_t* __restrict__ houtT,        // [2048][3008] bf16
                 float*  __restrict__ outp,         // [1000][2048] f32
                 int first) {
    __shared__ alignas(64) SMem sm;

    const int t    = threadIdx.x;
    const int lane = t & 31;
    const int wv   = t >> 5;        // wave 0..3
    const int m16  = lane & 15;
    const int kh   = lane >> 4;     // lane-half -> K group
    const int wm   = wv >> 1, wn = wv & 1;
    const int mbase = blockIdx.y * 64;
    const int nbase = blockIdx.x * 64;

    // ---- async staging setup: each thread owns 2 A-octets + 2 B-octets (16 B each) ----
    const int qr = t >> 2, qo = t & 3;                       // A: row, K-octet
    const int qp = qo;                                       // identity: octets kept in natural K order
    const int uc = t >> 2, uq = t & 3;                       // B: col, K-octet

    unsigned avo0 = (unsigned)(((mbase + qr) * KPAD + 8 * qo) * 2);
    unsigned avo1 = avo0 + (unsigned)(32 * KPAD * 2);
    unsigned bvo0 = (unsigned)(((nbase + uc) * MPAD + 8 * uq) * 2);
    unsigned bvo1 = bvo0 + (unsigned)(32 * MPAD * 2);

    const unsigned adst0 = (unsigned)(uintptr_t)&sm.s.A[0][qr][qp * 8];
    const unsigned bdst0 = (unsigned)(uintptr_t)&sm.s.Bt[0][uc][uq * 8];
    const unsigned long long abase = (unsigned long long)(uintptr_t)adjb;
    const unsigned long long bbase = (unsigned long long)(uintptr_t)hT;

    auto issue = [&](int buf) {
        const unsigned ad = adst0 + (buf ? 4096u : 0u);
        const unsigned bd = bdst0 + (buf ? 4096u : 0u);
        asm volatile("global_load_async_to_lds_b128 %0, %1, %2"
                     :: "v"(ad),          "v"(avo0), "s"(abase) : "memory");
        asm volatile("global_load_async_to_lds_b128 %0, %1, %2"
                     :: "v"(ad + 2048u),  "v"(avo1), "s"(abase) : "memory");
        asm volatile("global_load_async_to_lds_b128 %0, %1, %2"
                     :: "v"(bd),          "v"(bvo0), "s"(bbase) : "memory");
        asm volatile("global_load_async_to_lds_b128 %0, %1, %2"
                     :: "v"(bd + 2048u),  "v"(bvo1), "s"(bbase) : "memory");
        avo0 += 64u; avo1 += 64u; bvo0 += 64u; bvo1 += 64u;
    };

    v8f acc[2][2] = {};
    issue(0);

    for (int i = 0; i < NKSTEPS; ++i) {
        const int cur = i & 1;
        if (i + 1 < NKSTEPS) {
            issue(cur ^ 1);                                   // prefetch next tile
            asm volatile("s_wait_asynccnt 0x4" ::: "memory"); // current tile landed
        } else {
            asm volatile("s_wait_asynccnt 0x0" ::: "memory");
        }
        __syncthreads();

        const bf16_t (*Ab)[32] = sm.s.A[cur];
        const bf16_t (*Bb)[32] = sm.s.Bt[cur];

        v16bf af[2], bfr[2];
        #pragma unroll
        for (int mi = 0; mi < 2; ++mi)
            af[mi] = *(const v16bf*)&Ab[wm * 32 + mi * 16 + m16][kh * 16];
        #pragma unroll
        for (int ni = 0; ni < 2; ++ni)
            bfr[ni] = *(const v16bf*)&Bb[wn * 32 + ni * 16 + m16][kh * 16];

        #pragma unroll
        for (int mi = 0; mi < 2; ++mi)
            #pragma unroll
            for (int ni = 0; ni < 2; ++ni)
                acc[mi][ni] = __builtin_amdgcn_wmma_f32_16x16x32_bf16(
                    false, af[mi], false, bfr[ni], (short)0, acc[mi][ni], false, false);

        __syncthreads();
    }

    // ---- spill aggregate tile to LDS (C/D layout: VGPR r -> M = kh*8+r) ----
    #pragma unroll
    for (int mi = 0; mi < 2; ++mi)
        #pragma unroll
        for (int ni = 0; ni < 2; ++ni) {
            const int col = wn * 32 + ni * 16 + m16;
            #pragma unroll
            for (int r = 0; r < 8; ++r)
                sm.H[wm * 32 + mi * 16 + kh * 8 + r][col] = acc[mi][ni][r];
        }
    __syncthreads();

    // ---- fused lr = H(64x64) @ W(64x128) via WMMA; wave wv owns rows 16*wv..+15 ----
    v8f lr[8] = {};
    const int arow = wv * 16 + m16;
    #pragma unroll
    for (int ks = 0; ks < 2; ++ks) {
        v16bf afe;
        #pragma unroll
        for (int e = 0; e < 8; ++e) {
            afe[e]     = (bf16_t)sm.H[arow][ks * 32 + kh * 16 + e];
            afe[8 + e] = (bf16_t)sm.H[arow][ks * 32 + kh * 16 + 8 + e];
        }
        #pragma unroll
        for (int ni = 0; ni < 8; ++ni) {
            const v16bf bw = *(const v16bf*)&wt[(16 * ni + m16) * ODIM + ks * 32 + kh * 16];
            lr[ni] = __builtin_amdgcn_wmma_f32_16x16x32_bf16(
                false, afe, false, bw, (short)0, lr[ni], false, false);
        }
    }

    // ---- bias + GLU + packed transposed store + running max ----
    const int grow0 = mbase + wv * 16 + kh * 8;   // multiple of 8
    #pragma unroll
    for (int ni = 0; ni < 4; ++ni) {
        const int j  = 16 * ni + m16;
        const float bl = bias[j];
        const float br = bias[64 + j];
        alignas(16) bf16_t pk[8];
        float vals[8];
        #pragma unroll
        for (int r = 0; r < 8; ++r) {
            const float lhs = lr[ni][r] + bl;
            const float rhs = lr[ni + 4][r] + br;
            const float v = lhs * (1.0f / (1.0f + __expf(-rhs)));
            vals[r] = v;
            pk[r] = (bf16_t)v;
        }
        if (grow0 < NV) {
            *(uint4*)&houtT[(long long)(nbase + j) * MPAD + grow0] = *(const uint4*)pk;
            if (grow0 >= 1000 && grow0 < 2000) {
                float* o = outp + (long long)(grow0 - 1000) * NCOL + (nbase + j);
                #pragma unroll
                for (int r = 0; r < 8; ++r) {
                    float* p = o + (long long)r * NCOL;
                    *p = first ? vals[r] : fmaxf(*p, vals[r]);
                }
            }
        }
    }
}

// ---------------- host launch ----------------

extern "C" void kernel_launch(void* const* d_in, const int* in_sizes, int n_in,
                              void* d_out, int out_size, void* d_ws, size_t ws_size,
                              hipStream_t stream) {
    const float* x   = (const float*)d_in[0];
    const float* adj = (const float*)d_in[1];
    const float* W0  = (const float*)d_in[2];
    const float* b0  = (const float*)d_in[3];
    const float* W1  = (const float*)d_in[4];
    const float* b1  = (const float*)d_in[5];
    const float* W2  = (const float*)d_in[6];
    const float* b2  = (const float*)d_in[7];
    float* outp = (float*)d_out;

    char* ws = (char*)d_ws;
    const size_t szAdj = (size_t)MPAD * KPAD * sizeof(bf16_t);   // ~18.1 MB
    const size_t szH   = (size_t)NCOL * MPAD * sizeof(bf16_t);   // ~12.3 MB
    size_t off = 0;
    bf16_t* adjb = (bf16_t*)(ws + off); off = WS_ALIGN(off + szAdj);
    bf16_t* hA   = (bf16_t*)(ws + off); off = WS_ALIGN(off + szH);
    bf16_t* hB   = (bf16_t*)(ws + off); off = WS_ALIGN(off + szH);
    bf16_t* wt   = (bf16_t*)(ws + off);                          // 3*128*64 bf16

    {
        long long n = (long long)MPAD * KPAD;
        k_cvt_adj<<<(unsigned)((n + 255) / 256), 256, 0, stream>>>(adj, adjb);
    }
    k_cvt_xT<<<dim3(MPAD / 32, NCOL / 32), dim3(32, 8), 0, stream>>>(x, hA);
    k_cvt_w<<<(3 * 128 * ODIM + 255) / 256, 256, 0, stream>>>(W0, W1, W2, wt);
    k_zero_padT<<<(NCOL * (MPAD - NV) + 255) / 256, 256, 0, stream>>>(hB);

    dim3 grid(NCOL / 64, MPAD / 64);   // (32, 47)
    k_gcn_layer<<<grid, 128, 0, stream>>>(adjb, hA, wt + 0 * 128 * ODIM, b0, hB, outp, 1);
    k_gcn_layer<<<grid, 128, 0, stream>>>(adjb, hB, wt + 1 * 128 * ODIM, b1, hA, outp, 0);
    k_gcn_layer<<<grid, 128, 0, stream>>>(adjb, hA, wt + 2 * 128 * ODIM, b2, hB, outp, 0);
}
